// LSTMPlus_44341242364010
// MI455X (gfx1250) — matrix operand order repaired and checked by reading, not compiled, not run on hardware
//
#include <hip/hip_runtime.h>
#include <hip/hip_bf16.h>
#include <math.h>

typedef __attribute__((ext_vector_type(16))) __bf16 v16bf;
typedef __attribute__((ext_vector_type(8)))  __bf16 v8bf;
typedef __attribute__((ext_vector_type(8)))  float  v8f;

#define B_   64
#define T_   512
#define I_   128
#define H_   256
#define R_   20
#define DA_  50
#define C_   10
#define G4H  (4 * H_)     // 1024
#define KTOT (I_ + H_)    // 384 = fused K dim: [M_t | h]

__device__ __forceinline__ float sigf(float x) { return 1.0f / (1.0f + __expf(-x)); }

// ---------------------------------------------------------------------------
// Kernel 0: init — pack [W_ih; W_hh] into bf16 Wcat[KTOT][4H], h0 = c0 = 1,
// zero the grid-sync counter (fresh every call -> deterministic).
// ---------------------------------------------------------------------------
__global__ void init_kernel(const float* __restrict__ W_ih, const float* __restrict__ W_hh,
                            __bf16* __restrict__ Wcat, __bf16* __restrict__ hbf,
                            float* __restrict__ hf32, unsigned* __restrict__ cnt) {
    int idx = blockIdx.x * blockDim.x + threadIdx.x;
    if (idx < KTOT * G4H) {
        int k = idx / G4H, n = idx - k * G4H;
        float v = (k < I_) ? W_ih[k * G4H + n] : W_hh[(k - I_) * G4H + n];
        Wcat[idx] = (__bf16)v;
    }
    if (idx < B_ * H_) { hbf[idx] = (__bf16)1.0f; hf32[idx] = 1.0f; }
    if (idx == 0) *cnt = 0u;
}

// ---------------------------------------------------------------------------
// Kernel 1: attention scores S2[b,t,r] = tanh(x·W1^T + b1)·W2^T + b2
// One block per (b,t); memory-bound, VALU.
// ---------------------------------------------------------------------------
__global__ __launch_bounds__(64)
void attn_kernel(const float* __restrict__ x, const float* __restrict__ W1,
                 const float* __restrict__ b1, const float* __restrict__ W2,
                 const float* __restrict__ b2, float* __restrict__ S2) {
    __shared__ float xrow[I_];
    __shared__ float srow[DA_];
    const int blk = blockIdx.x;   // b*T + t
    const int tid = threadIdx.x;
    const float* xp = x + (size_t)blk * I_;
    xrow[tid]      = xp[tid];
    xrow[tid + 64] = xp[tid + 64];
    __syncthreads();
    if (tid < DA_) {
        float acc = b1[tid];
        const float* wr = W1 + tid * I_;
        #pragma unroll 4
        for (int i = 0; i < I_; ++i) acc += xrow[i] * wr[i];
        srow[tid] = tanhf(acc);
    }
    __syncthreads();
    if (tid < R_) {
        float acc = b2[tid];
        const float* wr = W2 + tid * DA_;
        #pragma unroll 5
        for (int d = 0; d < DA_; ++d) acc += srow[d] * wr[d];
        S2[(size_t)blk * R_ + tid] = acc;
    }
}

// ---------------------------------------------------------------------------
// Kernel 2: prefix-softmax mixing.  One block per batch b; 20 waves (one per
// r) keep running denom (scalar) + numer (4 floats/lane = 128) in registers,
// reduce over r through LDS, write Mbf[b,t,:] directly in bf16 (the LSTM's
// WMMA A-operand precision — identical RNE cast, done once instead of
// 16x per step inside the serial recurrence).
// ---------------------------------------------------------------------------
__global__ __launch_bounds__(640)
void attnmix_kernel(const float* __restrict__ x, const float* __restrict__ S2,
                    __bf16* __restrict__ Mbf) {
    __shared__ float cbuf[R_ * I_];
    const int b = blockIdx.x, tid = threadIdx.x;
    const int w = tid >> 5, lane = tid & 31;      // w == r (blockDim = 20 waves)
    const float* s2b = S2 + (size_t)b * T_ * R_ + w;
    // global max over T for this (b, r)
    float mx = -INFINITY;
    for (int t = lane; t < T_; t += 32) mx = fmaxf(mx, s2b[t * R_]);
    #pragma unroll
    for (int off = 16; off > 0; off >>= 1) mx = fmaxf(mx, __shfl_xor(mx, off, 32));

    float  denom = 0.0f;
    float4 numer = make_float4(0.f, 0.f, 0.f, 0.f);
    const float* xb = x + (size_t)b * T_ * I_;
    __bf16* Mb = Mbf + (size_t)b * T_ * I_;

    for (int t = 0; t < T_; ++t) {
        float e = __expf(s2b[t * R_] - mx);
        denom += e;
        float4 xv = *(const float4*)(xb + t * I_ + lane * 4);
        numer.x += e * xv.x; numer.y += e * xv.y;
        numer.z += e * xv.z; numer.w += e * xv.w;
        float inv = 1.0f / denom;
        float4 ct = make_float4(numer.x * inv, numer.y * inv, numer.z * inv, numer.w * inv);
        *(float4*)(cbuf + w * I_ + lane * 4) = ct;
        __syncthreads();
        if (tid < I_) {
            float s = 0.0f;
            #pragma unroll 4
            for (int rr = 0; rr < R_; ++rr) s += cbuf[rr * I_ + tid];
            Mb[t * I_ + tid] = (__bf16)(s * (1.0f / R_));
        }
        __syncthreads();
    }
}

// ---------------------------------------------------------------------------
// Kernel 3: fused LSTM recurrence with WMMA.
//   gates[64x1024] = [M_t | h] (64x384, bf16) x Wcat (384x1024, bf16), f32 acc.
//   Grid: 16 blocks (n-tile = 16 hidden cols) x 4 waves (m-tile = 16 batch).
//   Each wave computes all 4 gates for its 16x16 block -> c stays in VGPRs.
//   Weight slice staged once in LDS as [col][k] so each B fragment is one
//   contiguous 32B ds load.  h exchanged through L2 + grid spin barrier.
// ---------------------------------------------------------------------------
__global__ __launch_bounds__(128)
void lstm_kernel(const __bf16* __restrict__ Mbf, const float* __restrict__ bias,
                 const __bf16* __restrict__ Wcat,
                 __bf16* __restrict__ hbf, float* __restrict__ hf32,
                 unsigned* __restrict__ cnt) {
    __shared__ __bf16 Wlds[64 * KTOT];   // 48 KB: [col64 = g*16+c][k]
    const int nt   = blockIdx.x;         // 0..15
    const int tid  = threadIdx.x;
    const int w    = tid >> 5;           // m-tile 0..3
    const int lane = tid & 31;
    const int n0   = nt * 16;

    // stage this block's weight slice (4 gates x 16 cols x 384 k)
    for (int idx = tid; idx < 64 * KTOT; idx += 128) {
        int col = idx / KTOT, k = idx - col * KTOT;
        int g = col >> 4, c = col & 15;
        Wlds[idx] = Wcat[(size_t)k * G4H + g * H_ + n0 + c];
    }
    __syncthreads();

    const int half = lane >> 4;
    const int arow = w * 16 + (lane & 15);   // A-matrix row this lane feeds
    const int ncol = lane & 15;              // C/D column this lane owns

    const float bi  = bias[0 * H_ + n0 + ncol];
    const float bf_ = bias[1 * H_ + n0 + ncol];
    const float bg  = bias[2 * H_ + n0 + ncol];
    const float bo  = bias[3 * H_ + n0 + ncol];

    v8f cst;                                  // cell state tile, c0 = 1
    #pragma unroll
    for (int e = 0; e < 8; ++e) cst[e] = 1.0f;

    const __bf16* Mbase = Mbf + (size_t)arow * T_ * I_;

    for (int t = 0; t < T_; ++t) {
        // ---- A fragments: K 0..127 from Mbf, 128..383 from h (all bf16) ----
        v16bf afrag[12];
        const __bf16* Mrow = Mbase + (size_t)t * I_;
        #pragma unroll
        for (int j = 0; j < 4; ++j) {
            int kb = j * 32 + half * 16;
            v8bf m0 = *(const v8bf*)(Mrow + kb);
            v8bf m1 = *(const v8bf*)(Mrow + kb + 8);
            v16bf a;
            #pragma unroll
            for (int e = 0; e < 8; ++e) { a[e] = m0[e]; a[e + 8] = m1[e]; }
            afrag[j] = a;
        }
        const __bf16* hrow = hbf + arow * H_;
        #pragma unroll
        for (int j = 4; j < 12; ++j) {
            int kb = (j - 4) * 32 + half * 16;
            v8bf h0 = *(const v8bf*)(hrow + kb);
            v8bf h1 = *(const v8bf*)(hrow + kb + 8);
            v16bf a;
            #pragma unroll
            for (int e = 0; e < 8; ++e) { a[e] = h0[e]; a[e + 8] = h1[e]; }
            afrag[j] = a;
        }

        // prefetch next step's M row while this step computes / syncs
        if (t + 1 < T_)
            __builtin_prefetch((const void*)(Mbase + (size_t)(t + 1) * I_), 0, 0);

        // ---- 4 gates x 12 k-tiles of v_wmma_f32_16x16x32_bf16 ----
        v8f acc[4];
        #pragma unroll
        for (int g = 0; g < 4; ++g) {
            #pragma unroll
            for (int e = 0; e < 8; ++e) acc[g][e] = 0.0f;
        }
        #pragma unroll
        for (int g = 0; g < 4; ++g) {
            const __bf16* wp = &Wlds[(g * 16 + ncol) * KTOT + half * 16];
            #pragma unroll
            for (int j = 0; j < 12; ++j) {
                v16bf bfrag = *(const v16bf*)(wp + j * 32);
                acc[g] = __builtin_amdgcn_wmma_f32_16x16x32_bf16(
                             false, afrag[j], false, bfrag,
                             (short)0, acc[g], false, false);
            }
        }

        // ---- gate nonlinearities, cell/hidden update ----
        #pragma unroll
        for (int e = 0; e < 8; ++e) {
            float iv = sigf(acc[0][e] + bi);
            float fv = sigf(acc[1][e] + bf_);
            float gv = tanhf(acc[2][e] + bg);
            float ov = sigf(acc[3][e] + bo);
            float cv = fv * cst[e] + iv * gv;
            cst[e] = cv;
            float hv = ov * tanhf(cv);
            int row = w * 16 + e + 8 * half;            // C/D layout row
            hbf [row * H_ + n0 + ncol] = (__bf16)hv;
            hf32[row * H_ + n0 + ncol] = hv;
        }

        // ---- grid-wide step barrier (16 co-resident blocks) ----
        __threadfence();
        __syncthreads();
        if (tid == 0) {
            __hip_atomic_fetch_add(cnt, 1u, __ATOMIC_RELEASE, __HIP_MEMORY_SCOPE_AGENT);
            unsigned target = (unsigned)(t + 1) * gridDim.x;
            while (__hip_atomic_load(cnt, __ATOMIC_ACQUIRE, __HIP_MEMORY_SCOPE_AGENT) < target)
                __builtin_amdgcn_s_sleep(1);
        }
        __syncthreads();
    }
}

// ---------------------------------------------------------------------------
// Kernel 4: logits + softmax.  One thread per batch row.
// ---------------------------------------------------------------------------
__global__ __launch_bounds__(64)
void head_kernel(const float* __restrict__ hf32, const float* __restrict__ Wfc,
                 const float* __restrict__ bfc, float* __restrict__ out) {
    const int b = threadIdx.x;
    const float* h = hf32 + b * H_;
    float logits[C_];
    float mx = -INFINITY;
    #pragma unroll
    for (int c = 0; c < C_; ++c) {
        float acc = bfc[c];
        const float* wr = Wfc + c * H_;
        for (int j = 0; j < H_; ++j) acc += h[j] * wr[j];
        logits[c] = acc;
        mx = fmaxf(mx, acc);
    }
    float sum = 0.0f;
    #pragma unroll
    for (int c = 0; c < C_; ++c) { logits[c] = __expf(logits[c] - mx); sum += logits[c]; }
    float inv = 1.0f / sum;
    #pragma unroll
    for (int c = 0; c < C_; ++c) out[b * C_ + c] = logits[c] * inv;
}

// ---------------------------------------------------------------------------
extern "C" void kernel_launch(void* const* d_in, const int* in_sizes, int n_in,
                              void* d_out, int out_size, void* d_ws, size_t ws_size,
                              hipStream_t stream) {
    const float* x    = (const float*)d_in[0];
    const float* W_ih = (const float*)d_in[1];
    const float* W_hh = (const float*)d_in[2];
    const float* bb   = (const float*)d_in[3];
    const float* W1   = (const float*)d_in[4];
    const float* b1   = (const float*)d_in[5];
    const float* W2   = (const float*)d_in[6];
    const float* b2   = (const float*)d_in[7];
    const float* Wfc  = (const float*)d_in[8];
    const float* bfc  = (const float*)d_in[9];
    float* out = (float*)d_out;

    char* ws = (char*)d_ws;
    size_t off = 0;
    float*    S2   = (float*)(ws + off);    off += (size_t)B_ * T_ * R_ * 4;  // 2.62 MB
    __bf16*   Mbf  = (__bf16*)(ws + off);   off += (size_t)B_ * T_ * I_ * 2;  // 8.39 MB
    __bf16*   Wcat = (__bf16*)(ws + off);   off += (size_t)KTOT * G4H * 2;    // 0.79 MB
    __bf16*   hbf  = (__bf16*)(ws + off);   off += (size_t)B_ * H_ * 2;
    float*    hf32 = (float*)(ws + off);    off += (size_t)B_ * H_ * 4;
    unsigned* cnt  = (unsigned*)(ws + off); off += 256;

    init_kernel<<<(KTOT * G4H + 255) / 256, 256, 0, stream>>>(W_ih, W_hh, Wcat, hbf, hf32, cnt);
    attn_kernel<<<B_ * T_, 64, 0, stream>>>(x, W1, b1, W2, b2, S2);
    attnmix_kernel<<<B_, 640, 0, stream>>>(x, S2, Mbf);
    lstm_kernel<<<16, 128, 0, stream>>>(Mbf, bb, Wcat, hbf, hf32, cnt);
    head_kernel<<<1, 64, 0, stream>>>(hf32, Wfc, bfc, out);
}
